// MultiHeadAttention_71227737637117
// MI455X (gfx1250) — matrix operand order repaired and checked by reading, not compiled, not run on hardware
//
#include <hip/hip_runtime.h>

// ---------------------------------------------------------------------------
// MultiHeadAttention forward for MI455X (gfx1250, wave32, WMMA).
// N=4, L=2048, D=512, H=8, DK=64.
// Pipeline:
//   1) cvt_w:      Wq/Wk/Wv/Wo f32 -> f16 in workspace
//   2) gemm_bias:  Q/K = x @ W.T + b   (f32 A on the fly -> f16 out, WMMA;
//                  W tile double-buffered in LDS via global_load_async_to_lds)
//                  V   = x @ Wv.T + bv stored TRANSPOSED per head:
//                        Vt[n][h][dk][j]  (so attention needs no V transpose)
//   3) attn:       flash-attention per (n,h); causal + padding; f32 accum;
//                  per-wave causal early exit; no block barriers.
//   4) gemm_bias:  out = attn @ Wo.T + bo  (f16 A -> f32 out = d_out)
// Workspace: 4*512*512 f16 weights + 4 * 8192*512 f16 tensors = ~36 MB.
// ---------------------------------------------------------------------------

typedef __attribute__((ext_vector_type(16))) _Float16 v16h;
typedef __attribute__((ext_vector_type(8)))  _Float16 v8h;
typedef __attribute__((ext_vector_type(4)))  _Float16 v4h;
typedef __attribute__((ext_vector_type(8)))  float    v8f;
typedef __attribute__((ext_vector_type(4)))  float    v4f;

#define NBATCH 4
#define SEQL   2048
#define DMODEL 512
#define NHEAD  8
#define DHEAD  64

// --- WMMA wrapper: D = A(16x32 f16) * B(32x16 f16) + C(16x16 f32) ----------
static __device__ __forceinline__ v8f wmma32(v16h a, v16h b, v8f c) {
  return __builtin_amdgcn_wmma_f32_16x16x32_f16(
      /*neg_a=*/false, a, /*neg_b=*/false, b,
      /*c_mod=*/(short)0, c, /*reuse_a=*/false, /*reuse_b=*/false);
}

// --- Fragment loaders ------------------------------------------------------
// A-matrix 16x32 f16 layout (ISA 7.12.2): lane m = lane&15.
//   lane<16 : halves 0..7 = K 0..7,  halves 8..15 = K 16..23
//   lane>=16: halves 0..7 = K 8..15, halves 8..15 = K 24..31
// => per lane: chunk at (k0+kb) and (k0+kb+16), kb = (lane<16)?0:8.
static __device__ __forceinline__ v16h load_a_f16(const _Float16* p) {
  v8h lo = *(const v8h*)p;
  v8h hi = *(const v8h*)(p + 16);
  v16h r;
#pragma unroll
  for (int i = 0; i < 8; ++i) { r[i] = lo[i]; r[8 + i] = hi[i]; }
  return r;
}

static __device__ __forceinline__ v16h load_a_f32(const float* p) {
  v4f a0 = *(const v4f*)(p + 0);
  v4f a1 = *(const v4f*)(p + 4);
  v4f b0 = *(const v4f*)(p + 16);
  v4f b1 = *(const v4f*)(p + 20);
  v16h r;
#pragma unroll
  for (int i = 0; i < 4; ++i) {
    r[i]      = (_Float16)a0[i];
    r[4 + i]  = (_Float16)a1[i];
    r[8 + i]  = (_Float16)b0[i];
    r[12 + i] = (_Float16)b1[i];
  }
  return r;
}

// B-matrix 32x16 f16 layout: lane holds column n = lane&15.
//   lane<16 : halves 0..7 = K 0..7, 8..15 = K 16..23; lane>=16: K 8..15, K 24..31.
// => chunks at (k0 + kb2) and (k0 + kb2 + 16), kb2 = (lane<16)?0:8.
static __device__ __forceinline__ v16h load_b_f16(const _Float16* p) {
  v8h lo = *(const v8h*)p;
  v8h hi = *(const v8h*)(p + 16);
  v16h r;
#pragma unroll
  for (int i = 0; i < 8; ++i) { r[i] = lo[i]; r[8 + i] = hi[i]; }
  return r;
}

// Async global->LDS lane copy: 32B (two b128, INST_OFFSET applies to both the
// LDS and global side per CDNA5 ISA 08_async_tensor 4.4). Tracked by ASYNCcnt.
static __device__ __forceinline__ void async_copy_b256(unsigned lds_off,
                                                       const _Float16* g) {
  asm volatile(
      "global_load_async_to_lds_b128 %0, %1, off\n\t"
      "global_load_async_to_lds_b128 %0, %1, off offset:16"
      :
      : "v"(lds_off), "v"((unsigned long long)(uintptr_t)g)
      : "memory");
}

// --- 1) weight conversion f32 -> f16 ---------------------------------------
__global__ void cvt_w_kernel(const float* __restrict__ src,
                             _Float16* __restrict__ dst, int n) {
  int i = (blockIdx.x * blockDim.x + threadIdx.x) * 4;
  if (i + 3 < n) {
    v4f x = *(const v4f*)(src + i);
    v4h y;
#pragma unroll
    for (int t = 0; t < 4; ++t) y[t] = (_Float16)x[t];
    *(v4h*)(dst + i) = y;
  }
}

// --- 2/4) GEMM with bias: Y[M,512] = A[M,512] @ W[512,512]^T + b -----------
// Block 256 threads = 8 waves; tile 128 rows x 64 cols; wave owns 16 rows.
// W tile is staged in LDS with double-buffered async global->LDS copies
// (64 cols x 64 K per chunk = 8 KB, 2 async b128 per thread per chunk).
// OUT_MODE: 0 = f16 row-major, 1 = f32 row-major, 2 = f16 transposed-per-head
//           (Vt[n][h][dk][j], contiguous along j == contiguous along e).
template <bool A_IS_F16, int OUT_MODE>
__global__ __launch_bounds__(256, 2)
void gemm_bias_kernel(const void* __restrict__ Ain,
                      const _Float16* __restrict__ W,
                      const float* __restrict__ bias,
                      void* __restrict__ Yout) {
  __shared__ _Float16 ldsW[2][64 * 64];        // double-buffered tile, 16 KB

  const int lane = threadIdx.x & 31;
  const int wv   = threadIdx.x >> 5;
  const int ln   = lane & 15;
  const int half = lane >> 4;

  const int row0 = blockIdx.y * 128 + wv * 16;
  const int col0 = blockIdx.x * 64;
  const int m    = row0 + ln;
  const int kbA  = half ? 8 : 0;
  const int kbB  = half ? 8 : 0;

  // Per-thread slice of the async W stage: 32B along K of one column.
  const int tcol = threadIdx.x >> 2;           // 0..63
  const int tk   = (threadIdx.x & 3) * 16;     // 0,16,32,48

  v8f acc[4];
#pragma unroll
  for (int a = 0; a < 4; ++a)
#pragma unroll
    for (int e = 0; e < 8; ++e) acc[a][e] = 0.0f;

  const float*    arow32 = A_IS_F16 ? nullptr
                         : (const float*)Ain + (size_t)m * DMODEL;
  const _Float16* arow16 = A_IS_F16
                         ? (const _Float16*)Ain + (size_t)m * DMODEL : nullptr;

  // Stage chunk 0.
  {
    const _Float16* g = W + (size_t)(col0 + tcol) * DMODEL + tk;
    async_copy_b256((unsigned)(uintptr_t)&ldsW[0][tcol * 64 + tk], g);
  }

  for (int c = 0; c < 8; ++c) {                // 8 chunks of K=64
    if (c < 7) {
      const _Float16* g =
          W + (size_t)(col0 + tcol) * DMODEL + (c + 1) * 64 + tk;
      async_copy_b256((unsigned)(uintptr_t)&ldsW[(c + 1) & 1][tcol * 64 + tk],
                      g);
      // 2 async instructions still in flight (next chunk); current done.
      asm volatile("s_wait_asynccnt 0x2" ::: "memory");
    } else {
      asm volatile("s_wait_asynccnt 0x0" ::: "memory");
    }
    __syncthreads();                           // publish chunk c to all waves

    const _Float16* wb = &ldsW[c & 1][0];
#pragma unroll
    for (int s2 = 0; s2 < 2; ++s2) {
      const int k0 = c * 64 + s2 * 32;
      v16h af = A_IS_F16 ? load_a_f16(arow16 + k0 + kbA)
                         : load_a_f32(arow32 + k0 + kbA);
#pragma unroll
      for (int a = 0; a < 4; ++a) {
        v16h bf = load_b_f16(wb + (a * 16 + ln) * 64 + s2 * 32 + kbB);
        acc[a] = wmma32(af, bf, acc[a]);
      }
    }
    __syncthreads();                           // done reading before overwrite
  }

#pragma unroll
  for (int a = 0; a < 4; ++a) {
    const int col = col0 + a * 16 + ln;
    const float bsc = bias[col];
    if (OUT_MODE == 2) {
      // transposed per-head store: base + (row & 2047) contiguous in e
      const int nn = row0 >> 11;               // batch (row0..row0+15 same n)
      const int hh = col >> 6;
      const int dk = col & 63;
      _Float16* dst = (_Float16*)Yout +
          (((size_t)nn * NHEAD + hh) * DHEAD + dk) * SEQL +
          ((row0 + half * 8) & (SEQL - 1));
#pragma unroll
      for (int e = 0; e < 8; ++e)
        dst[e] = (_Float16)(acc[a][e] + bsc);
    } else {
#pragma unroll
      for (int e = 0; e < 8; ++e) {
        const int row = row0 + half * 8 + e;
        const float v = acc[a][e] + bsc;
        if (OUT_MODE == 1)
          ((float*)Yout)[(size_t)row * DMODEL + col] = v;
        else
          ((_Float16*)Yout)[(size_t)row * DMODEL + col] = (_Float16)v;
      }
    }
  }
}

// --- 3) flash attention ----------------------------------------------------
// Grid: (L/128, N*H). Block 256 = 8 independent waves; wave owns 16 rows.
// No block barriers: K and Vt B-fragments load straight from global (both
// are contiguous along k: two 16B chunks per lane), so each wave runs to its own causal bound.
__global__ __launch_bounds__(256, 2)
void attn_kernel(const _Float16* __restrict__ Q,
                 const _Float16* __restrict__ K,
                 const _Float16* __restrict__ Vt,
                 const int* __restrict__ pad,
                 _Float16* __restrict__ Aout) {
  __shared__ _Float16 ldsP[8 * 16 * 64];       // per-wave P tiles, 16 KB

  const int lane = threadIdx.x & 31;
  const int wv   = threadIdx.x >> 5;
  const int ln   = lane & 15;
  const int half = lane >> 4;

  const int nh = blockIdx.y;
  const int n  = nh / NHEAD;
  const int h  = nh % NHEAD;
  const int qrow0 = blockIdx.x * 128 + wv * 16;
  const int rbase = half * 8;                  // rows this lane's C-regs hold
  const int kbA = half ? 8 : 0;
  const int kbB = half ? 8 : 0;

  // Q fragments (A-layout), 2 k-steps over DK=64, kept in registers.
  const _Float16* qrow =
      Q + (size_t)(n * SEQL + qrow0 + ln) * DMODEL + h * DHEAD;
  v16h qa[2];
#pragma unroll
  for (int s = 0; s < 2; ++s) qa[s] = load_a_f16(qrow + s * 32 + kbA);

  // Per-head transposed V base: Vt[n][h][dk][j]
  const _Float16* vbase = Vt + ((size_t)(n * NHEAD + h)) * DHEAD * SEQL;

  v8f o[4];
#pragma unroll
  for (int a = 0; a < 4; ++a)
#pragma unroll
    for (int e = 0; e < 8; ++e) o[a][e] = 0.0f;
  float m_i[8], l_i[8];
#pragma unroll
  for (int e = 0; e < 8; ++e) { m_i[e] = -1e30f; l_i[e] = 0.0f; }

  const int jmaxw = qrow0 + 16;                // per-wave causal bound

  for (int j0 = 0; j0 < jmaxw; j0 += 64) {
    // Prefetch next key block (global_prefetch_b8 on gfx1250).
    if (j0 + 64 < jmaxw) {
      __builtin_prefetch(
          K + (size_t)(n * SEQL + j0 + 64 + lane) * DMODEL + h * DHEAD, 0, 1);
      __builtin_prefetch(vbase + (size_t)lane * SEQL + j0 + 64, 0, 1);
    }

    // S = Q * K^T : B-fragments straight from global (K rows contiguous).
    v8f sc[4];
#pragma unroll
    for (int a = 0; a < 4; ++a) {
      const int jc = j0 + a * 16 + ln;
      const _Float16* kp =
          K + (size_t)(n * SEQL + jc) * DMODEL + h * DHEAD;
      v8f cc;
#pragma unroll
      for (int e = 0; e < 8; ++e) cc[e] = 0.0f;
#pragma unroll
      for (int s = 0; s < 2; ++s)
        cc = wmma32(qa[s], load_b_f16(kp + s * 32 + kbB), cc);
      sc[a] = cc;
    }

    // scale + causal + padding masks, per-row block max
    float mloc[8];
#pragma unroll
    for (int e = 0; e < 8; ++e) mloc[e] = -1e30f;
#pragma unroll
    for (int a = 0; a < 4; ++a) {
      const int jc = j0 + a * 16 + ln;
      const int pv = pad[n * SEQL + jc];
#pragma unroll
      for (int e = 0; e < 8; ++e) {
        const int row = qrow0 + rbase + e;
        float sv = sc[a][e] * 0.125f;             // 1/sqrt(64)
        const bool ok = (jc <= row) && (pv != 0);
        sv = ok ? sv : -1e30f;
        sc[a][e] = sv;
        mloc[e] = fmaxf(mloc[e], sv);
      }
    }
#pragma unroll
    for (int off = 1; off < 16; off <<= 1)
#pragma unroll
      for (int e = 0; e < 8; ++e)
        mloc[e] = fmaxf(mloc[e], __shfl_xor(mloc[e], off, 32));

    float alpha[8], rsum[8];
#pragma unroll
    for (int e = 0; e < 8; ++e) {
      const float mn = fmaxf(m_i[e], mloc[e]);
      alpha[e] = __expf(m_i[e] - mn);
      m_i[e] = mn;
      rsum[e] = 0.0f;
    }

    // P = exp(S - m), store f16 into wave-private LDS tile (C->A relayout).
    _Float16* pP = ldsP + wv * 1024;
#pragma unroll
    for (int a = 0; a < 4; ++a) {
      const int jl = a * 16 + ln;
#pragma unroll
      for (int e = 0; e < 8; ++e) {
        const float sv = sc[a][e];
        const float p = (sv <= -1e29f) ? 0.0f : __expf(sv - m_i[e]);
        rsum[e] += p;
        pP[(rbase + e) * 64 + jl] = (_Float16)p;
      }
    }
#pragma unroll
    for (int off = 1; off < 16; off <<= 1)
#pragma unroll
      for (int e = 0; e < 8; ++e)
        rsum[e] += __shfl_xor(rsum[e], off, 32);
#pragma unroll
    for (int e = 0; e < 8; ++e) l_i[e] = l_i[e] * alpha[e] + rsum[e];
#pragma unroll
    for (int a = 0; a < 4; ++a)
#pragma unroll
      for (int e = 0; e < 8; ++e) o[a][e] *= alpha[e];

    // Intra-wave LDS RAW fence before re-reading the P tile as A-fragments.
    asm volatile("s_wait_dscnt 0" ::: "memory");

    // O += P(16x64) * V(64x16 per n-tile); V^T fragments straight from
    // global (Vt is per-head [dk][j], contiguous along j).
#pragma unroll
    for (int s = 0; s < 2; ++s) {
      const _Float16* pr = ldsP + wv * 1024 + ln * 64 + s * 32;
      v16h pa = load_a_f16(pr + kbA);
#pragma unroll
      for (int a = 0; a < 4; ++a) {
        const _Float16* vtp =
            vbase + (size_t)(a * 16 + ln) * SEQL + j0 + s * 32 + kbB;
        o[a] = wmma32(pa, load_b_f16(vtp), o[a]);
      }
    }
  }

  // Normalize and write attention output (f16, [N*L, 512] head-concat).
  float inv[8];
#pragma unroll
  for (int e = 0; e < 8; ++e) inv[e] = 1.0f / l_i[e];
#pragma unroll
  for (int a = 0; a < 4; ++a) {
    const int col = h * DHEAD + a * 16 + ln;
#pragma unroll
    for (int e = 0; e < 8; ++e) {
      const int row = qrow0 + rbase + e;
      Aout[(size_t)(n * SEQL + row) * DMODEL + col] =
          (_Float16)(o[a][e] * inv[e]);
    }
  }
}

// ---------------------------------------------------------------------------
extern "C" void kernel_launch(void* const* d_in, const int* in_sizes, int n_in,
                              void* d_out, int out_size, void* d_ws,
                              size_t ws_size, hipStream_t stream) {
  (void)in_sizes; (void)n_in; (void)out_size; (void)ws_size;

  const float* x_q = (const float*)d_in[0];
  const float* x_k = (const float*)d_in[1];
  const float* x_v = (const float*)d_in[2];
  const int*   pad = (const int*)d_in[3];
  // d_in[4] attention_mask is the causal tril; applied analytically (jc<=row).
  const float* Wq = (const float*)d_in[5];
  const float* bq = (const float*)d_in[6];
  const float* Wk = (const float*)d_in[7];
  const float* bk = (const float*)d_in[8];
  const float* Wv = (const float*)d_in[9];
  const float* bv = (const float*)d_in[10];
  const float* Wo = (const float*)d_in[11];
  const float* bo = (const float*)d_in[12];

  const int WSZ = DMODEL * DMODEL;                     // 262144
  const size_t TSZ = (size_t)NBATCH * SEQL * DMODEL;   // 4194304

  _Float16* w16  = (_Float16*)d_ws;
  _Float16* Wq16 = w16;
  _Float16* Wk16 = Wq16 + WSZ;
  _Float16* Wv16 = Wk16 + WSZ;
  _Float16* Wo16 = Wv16 + WSZ;
  _Float16* Q16  = Wo16 + WSZ;
  _Float16* K16  = Q16 + TSZ;
  _Float16* Vt16 = K16 + TSZ;                          // transposed per head
  _Float16* A16  = Vt16 + TSZ;

  // 1) weights f32 -> f16
  {
    dim3 g(WSZ / (256 * 4));
    cvt_w_kernel<<<g, 256, 0, stream>>>(Wq, Wq16, WSZ);
    cvt_w_kernel<<<g, 256, 0, stream>>>(Wk, Wk16, WSZ);
    cvt_w_kernel<<<g, 256, 0, stream>>>(Wv, Wv16, WSZ);
    cvt_w_kernel<<<g, 256, 0, stream>>>(Wo, Wo16, WSZ);
  }

  // 2) Q/K/V projections (WMMA + async LDS W staging)
  dim3 gg(DMODEL / 64, (NBATCH * SEQL) / 128);  // (8, 64)
  gemm_bias_kernel<false, 0><<<gg, 256, 0, stream>>>(x_q, Wq16, bq, Q16);
  gemm_bias_kernel<false, 0><<<gg, 256, 0, stream>>>(x_k, Wk16, bk, K16);
  gemm_bias_kernel<false, 2><<<gg, 256, 0, stream>>>(x_v, Wv16, bv, Vt16);

  // 3) flash attention (per-wave causal early exit, no block barriers)
  dim3 ga(SEQL / 128, NBATCH * NHEAD);          // (16, 32)
  attn_kernel<<<ga, 256, 0, stream>>>(Q16, K16, Vt16, pad, A16);

  // 4) output projection -> f32 d_out
  gemm_bias_kernel<true, 1><<<gg, 256, 0, stream>>>(A16, Wo16, bo, d_out);
}
